// Multi_head_attention_45689862095015
// MI455X (gfx1250) — compile-verified, matrix operand order checked by reading, not run on hardware
//
#include <hip/hip_runtime.h>

#define DEV __device__ __forceinline__

typedef __bf16 bf16_t;
typedef __attribute__((ext_vector_type(16))) __bf16 v16bf;
typedef __attribute__((ext_vector_type(8)))  __bf16 v8bf;
typedef __attribute__((ext_vector_type(8)))  float  v8f;

static constexpr int Bc = 4, Tc = 1024, Cc = 768, NHc = 12, HSc = 64;

// ---------- helpers ----------

DEV bf16_t f2bf(float f) {
    unsigned u = __builtin_bit_cast(unsigned, f);
    unsigned r = (u + 0x7FFFu + ((u >> 16) & 1u)) >> 16;   // round-to-nearest-even
    unsigned short h = (unsigned short)r;
    return __builtin_bit_cast(bf16_t, h);
}

// Load a 16x32 bf16 fragment (A layout; also used for B via B^T stored N-major).
// ISA layout: lanes 0-15 hold row=lane with K-groups {0..7, 16..23};
//             lanes 16-31 hold row=lane-16 with K-groups {8..15, 24..31}.
DEV v16bf load_frag(const bf16_t* base, int ld) {
    int lane = threadIdx.x & 31;
    int row  = lane & 15;
    int kg   = (lane >> 4) << 3;            // 0 or 8
    const bf16_t* p = base + (size_t)row * ld + kg;
    v8bf lo = *(const v8bf*)(p);            // K = kg .. kg+7
    v8bf hi = *(const v8bf*)(p + 16);       // K = 16+kg .. 16+kg+7
    v16bf out;
#pragma unroll
    for (int i = 0; i < 8; ++i) { out[i] = lo[i]; out[i + 8] = hi[i]; }
    return out;
}

DEV v8f wmma_bf16(v16bf a, v16bf b, v8f c) {
    // (neg_a, A, neg_b, B, c_mod, C, reuse_a, reuse_b)
    return __builtin_amdgcn_wmma_f32_16x16x32_bf16(false, a, false, b, (short)0, c, false, false);
}

DEV float hmax16(float v) {
#pragma unroll
    for (int m = 1; m < 16; m <<= 1) v = fmaxf(v, __shfl_xor(v, m, 32));
    return v;
}
DEV float hsum16(float v) {
#pragma unroll
    for (int m = 1; m < 16; m <<= 1) v += __shfl_xor(v, m, 32);
    return v;
}

// CDNA5 async global->LDS copy: 16B per lane, tracked by ASYNCcnt.
// VDST gets the workgroup-relative LDS byte offset = low 32 bits of the
// generic shared-memory address (HW adds LDS_BASE). Passing the truncated
// pointer as an operand also CAPTURES the shared array, so the compiler
// keeps the subsequent ds_load reads live (asm memory clobber may write it).
DEV void async_copy_b128(const bf16_t* lds_ptr, const void* gaddr) {
    unsigned lds_off = (unsigned)(unsigned long long)lds_ptr;
    asm volatile("global_load_async_to_lds_b128 %0, %1, off"
                 :: "v"(lds_off), "v"(gaddr) : "memory");
}
#define WAIT_ASYNC_0() asm volatile("s_wait_asynccnt 0x0" ::: "memory")
#define WAIT_ASYNC_1() asm volatile("s_wait_asynccnt 0x1" ::: "memory")

// ---------- conversion kernels ----------

__global__ void cvt_kernel(const float* __restrict__ src, bf16_t* __restrict__ dst, int nelem) {
    int i = blockIdx.x * blockDim.x + threadIdx.x;
    int stride = gridDim.x * blockDim.x;
    for (; i < nelem; i += stride) dst[i] = f2bf(src[i]);
}

// W_[qkv]: [nh, C, hs] fp32 -> Wt: [3, nh, hs, C] bf16  (transposed to N x K)
__global__ void cvtw_kernel(const float* __restrict__ Wq, const float* __restrict__ Wk,
                            const float* __restrict__ Wv, bf16_t* __restrict__ Wt) {
    int w = blockIdx.y;
    const float* src = (w == 0) ? Wq : ((w == 1) ? Wk : Wv);
    const int total = NHc * Cc * HSc;
    int stride = gridDim.x * blockDim.x;
    for (int i = blockIdx.x * blockDim.x + threadIdx.x; i < total; i += stride) {
        int h   = i / (Cc * HSc);
        int rem = i - h * (Cc * HSc);
        int c   = rem / HSc;
        int d   = rem - c * HSc;
        Wt[((size_t)(w * NHc + h) * HSc + d) * Cc + c] = f2bf(src[i]);
    }
}

// ---------- QKV projection GEMM ----------
// Workgroup = 8 waves covering 128 rows x 64 cols for one (head, q/k/v).
// Weight tile (B^T, 64 x 32 per K-step, 4KB) is async-copied into LDS,
// double-buffered, shared by all 8 waves. A fragments stream from global
// with register prefetch.
__global__ __launch_bounds__(256) void qkv_kernel(
        const bf16_t* __restrict__ xb, const bf16_t* __restrict__ Wt,
        bf16_t* __restrict__ Qb, bf16_t* __restrict__ Kb, bf16_t* __restrict__ Vt) {
    int tid  = threadIdx.x;
    int wave = tid >> 5;
    int m0   = blockIdx.x * 128 + wave * 16;
    int h    = blockIdx.y;
    int w    = blockIdx.z;               // 0=Q 1=K 2=V
    int lane = tid & 31;
    int n    = lane & 15;
    int half = lane >> 4;

    const bf16_t* A  = xb + (size_t)m0 * Cc;
    const bf16_t* Bt = Wt + (size_t)(w * NHc + h) * HSc * Cc;

    // double-buffered B^T stage: [2][64 rows][32 bf16] = 2 x 4KB
    __shared__ __align__(16) bf16_t bstage[2][HSc * 32];

    // per-thread async-copy chunk: 256 threads x 16B = 4KB tile
    int d    = tid >> 2;                 // weight row (output dim), 0..63
    int part = tid & 3;                  // 4 x 8 bf16 chunks per row
    bf16_t* ldst0 = &bstage[0][d * 32 + part * 8];
    bf16_t* ldst1 = &bstage[1][d * 32 + part * 8];
    const bf16_t* gsrc = Bt + (size_t)d * Cc + part * 8;

    async_copy_b128(ldst0, gsrc);        // K-step 0 -> buffer 0

    v8f acc[4] = {{}, {}, {}, {}};
    v16bf a_c = load_frag(A, Cc);

    int buf = 0;
    for (int k = 0; k < Cc; k += 32, buf ^= 1) {
        bool have_next = (k + 32 < Cc);
        if (have_next) {
            // next tile into other buffer (that buffer's readers finished at the
            // end-of-iteration barrier of the previous step)
            async_copy_b128(buf == 0 ? ldst1 : ldst0, gsrc + k + 32);
            WAIT_ASYNC_1();              // current buffer's copy complete
        } else {
            WAIT_ASYNC_0();
        }
        __syncthreads();                 // all waves' copies visible

        v16bf a_n;
        if (have_next) a_n = load_frag(A + k + 32, Cc);   // overlap with ds reads

        const bf16_t* bp = &bstage[buf][0];
#pragma unroll
        for (int nf = 0; nf < 4; ++nf)
            acc[nf] = wmma_bf16(a_c, load_frag(bp + nf * 16 * 32, 32), acc[nf]);

        if (have_next) a_c = a_n;
        __syncthreads();                 // readers done before buffer is overwritten
    }

    int b  = m0 / Tc;
    int t0 = m0 % Tc;
    if (w < 2) {
        bf16_t* out = (w == 0 ? Qb : Kb) + ((size_t)(b * NHc + h) * Tc + t0) * HSc;
#pragma unroll
        for (int nf = 0; nf < 4; ++nf)
#pragma unroll
            for (int r = 0; r < 8; ++r)
                out[(size_t)(r + (half << 3)) * HSc + nf * 16 + n] = f2bf(acc[nf][r]);
    } else {
        bf16_t* out = Vt + (size_t)(b * NHc + h) * HSc * Tc;
#pragma unroll
        for (int nf = 0; nf < 4; ++nf)
#pragma unroll
            for (int r = 0; r < 8; ++r)
                out[(size_t)(nf * 16 + n) * Tc + t0 + r + (half << 3)] = f2bf(acc[nf][r]);
    }
}

// ---------- flash attention (one wave per 16-row Q block per head) ----------
// Software-pipelined: V fragments and next-step K fragments are issued before
// the softmax VALU burst so exp/shuffle work hides the load latency.
__global__ void attn_kernel(const bf16_t* __restrict__ Qb, const bf16_t* __restrict__ Kb,
                            const bf16_t* __restrict__ Vt, bf16_t* __restrict__ resb) {
    const float scale = 0.03125f;        // 1/sqrt(context_length=1024), reference quirk
    int t0   = blockIdx.x << 4;
    int h    = blockIdx.y;
    int b    = blockIdx.z;
    int lane = threadIdx.x & 31;
    int n    = lane & 15;
    int half = lane >> 4;

    const bf16_t* Qbase = Qb + ((size_t)(b * NHc + h) * Tc + t0) * HSc;
    const bf16_t* Kbase = Kb + (size_t)(b * NHc + h) * Tc * HSc;
    const bf16_t* Vbase = Vt + (size_t)(b * NHc + h) * HSc * Tc;

    v16bf qa0 = load_frag(Qbase, HSc);        // head-dim K = 0..31
    v16bf qa1 = load_frag(Qbase + 32, HSc);   // head-dim K = 32..63

    v8f o0 = {}, o1 = {}, o2 = {}, o3 = {};
    float mrow[8], lrow[8];
#pragma unroll
    for (int r = 0; r < 8; ++r) { mrow[r] = -3.0e38f; lrow[r] = 0.0f; }

    __shared__ __align__(16) bf16_t plds[16 * 32];

    // preload K fragments for first 32-key tile
    v16bf kf0 = load_frag(Kbase, HSc);
    v16bf kf1 = load_frag(Kbase + 32, HSc);
    v16bf kf2 = load_frag(Kbase + (size_t)16 * HSc, HSc);
    v16bf kf3 = load_frag(Kbase + (size_t)16 * HSc + 32, HSc);

    for (int j0 = 0; j0 <= t0; j0 += 32) {
        v8f s0 = {}, s1 = {};
        s0 = wmma_bf16(qa0, kf0, s0);
        s0 = wmma_bf16(qa1, kf1, s0);
        s1 = wmma_bf16(qa0, kf2, s1);
        s1 = wmma_bf16(qa1, kf3, s1);

        // issue V fragment loads for this step early
        v16bf vf0 = load_frag(Vbase + (size_t)0  * Tc + j0, Tc);
        v16bf vf1 = load_frag(Vbase + (size_t)16 * Tc + j0, Tc);
        v16bf vf2 = load_frag(Vbase + (size_t)32 * Tc + j0, Tc);
        v16bf vf3 = load_frag(Vbase + (size_t)48 * Tc + j0, Tc);
        // issue next-tile K fragment loads; softmax below hides their latency
        if (j0 + 32 <= t0) {
            const bf16_t* kn = Kbase + (size_t)(j0 + 32) * HSc;
            kf0 = load_frag(kn, HSc);
            kf1 = load_frag(kn + 32, HSc);
            kf2 = load_frag(kn + (size_t)16 * HSc, HSc);
            kf3 = load_frag(kn + (size_t)16 * HSc + 32, HSc);
        }

        int key0 = j0 + n;
        int key1 = j0 + 16 + n;
#pragma unroll
        for (int r = 0; r < 8; ++r) {
            int row  = t0 + r + (half << 3);
            float a  = (key0 <= row) ? s0[r] * scale : -3.0e38f;
            float bb = (key1 <= row) ? s1[r] * scale : -3.0e38f;
            float vmax  = hmax16(fmaxf(a, bb));
            float nm    = fmaxf(mrow[r], vmax);
            float alpha = __expf(mrow[r] - nm);
            float p0    = __expf(a - nm);
            float p1    = __expf(bb - nm);
            mrow[r] = nm;
            lrow[r] = lrow[r] * alpha + hsum16(p0 + p1);
            o0[r] *= alpha; o1[r] *= alpha; o2[r] *= alpha; o3[r] *= alpha;
            int m = r + (half << 3);
            plds[m * 32 + n]      = f2bf(p0);   // D-layout -> row-major P tile in LDS
            plds[m * 32 + 16 + n] = f2bf(p1);
        }
        __syncthreads();
        v16bf pa = load_frag(plds, 32);         // P as A-fragment (16x32 over keys)
        o0 = wmma_bf16(pa, vf0, o0);
        o1 = wmma_bf16(pa, vf1, o1);
        o2 = wmma_bf16(pa, vf2, o2);
        o3 = wmma_bf16(pa, vf3, o3);
        __syncthreads();
    }

#pragma unroll
    for (int r = 0; r < 8; ++r) {
        float inv = 1.0f / lrow[r];
        int row = t0 + r + (half << 3);
        bf16_t* out = resb + (size_t)(b * Tc + row) * Cc + h * HSc;
        out[n]      = f2bf(o0[r] * inv);
        out[16 + n] = f2bf(o1[r] * inv);
        out[32 + n] = f2bf(o2[r] * inv);
        out[48 + n] = f2bf(o3[r] * inv);
    }
}

// ---------- output linear: y = res @ lin_w^T + b  (fp32 out) ----------
// Register double-buffered: next K-step fragments are in flight while the
// current WMMAs execute; prefetch hints pull the following tile into cache.
__global__ void linear_kernel(const bf16_t* __restrict__ resb, const bf16_t* __restrict__ lwb,
                              const float* __restrict__ bias, float* __restrict__ out) {
    int m0   = blockIdx.x << 4;
    int n0   = blockIdx.y << 6;
    int lane = threadIdx.x & 31;
    int n    = lane & 15;
    int half = lane >> 4;

    const bf16_t* A  = resb + (size_t)m0 * Cc;
    const bf16_t* Bt = lwb + (size_t)n0 * Cc;   // lin_w is [out, in] = N x K already

    v8f acc[4] = {{}, {}, {}, {}};
    v16bf a_c = load_frag(A, Cc);
    v16bf b_c[4];
#pragma unroll
    for (int nf = 0; nf < 4; ++nf) b_c[nf] = load_frag(Bt + (size_t)(nf * 16) * Cc, Cc);

    for (int k = 0; k < Cc; k += 32) {
        bool have_next = (k + 32 < Cc);
        v16bf a_n;
        v16bf b_n[4];
        if (have_next) {
            a_n = load_frag(A + k + 32, Cc);
#pragma unroll
            for (int nf = 0; nf < 4; ++nf)
                b_n[nf] = load_frag(Bt + (size_t)(nf * 16) * Cc + k + 32, Cc);
            if (k + 64 < Cc) __builtin_prefetch(Bt + k + 64, 0, 1);
        }
#pragma unroll
        for (int nf = 0; nf < 4; ++nf)
            acc[nf] = wmma_bf16(a_c, b_c[nf], acc[nf]);
        if (have_next) {
            a_c = a_n;
#pragma unroll
            for (int nf = 0; nf < 4; ++nf) b_c[nf] = b_n[nf];
        }
    }
#pragma unroll
    for (int nf = 0; nf < 4; ++nf) {
        float bv = bias[n0 + nf * 16 + n];
#pragma unroll
        for (int r = 0; r < 8; ++r)
            out[(size_t)(m0 + r + (half << 3)) * Cc + n0 + nf * 16 + n] = acc[nf][r] + bv;
    }
}

// ---------- launch ----------

extern "C" void kernel_launch(void* const* d_in, const int* in_sizes, int n_in,
                              void* d_out, int out_size, void* d_ws, size_t ws_size,
                              hipStream_t stream) {
    (void)in_sizes; (void)n_in; (void)out_size; (void)ws_size;

    const float* x  = (const float*)d_in[0];
    const float* Wq = (const float*)d_in[1];
    const float* Wk = (const float*)d_in[2];
    const float* Wv = (const float*)d_in[3];
    const float* lw = (const float*)d_in[4];
    const float* lb = (const float*)d_in[5];
    float* out = (float*)d_out;

    bf16_t* ws = (bf16_t*)d_ws;
    const size_t nBTC = (size_t)Bc * Tc * Cc;     // 3,145,728 elements
    const size_t nW   = (size_t)NHc * Cc * HSc;   // 589,824 elements
    bf16_t* xb   = ws;                    // x in bf16           [B*T, C]
    bf16_t* Wt   = xb + nBTC;             // transposed weights  [3, nh, hs, C]
    bf16_t* lwb  = Wt + 3 * nW;           // lin_w bf16          [C, C]
    bf16_t* Qb   = lwb + (size_t)Cc * Cc; // Q                   [B, nh, T, hs]
    bf16_t* Kb   = Qb + nBTC;             // K                   [B, nh, T, hs]
    bf16_t* Vt   = Kb + nBTC;             // V transposed        [B, nh, hs, T]
    bf16_t* resb = Vt + nBTC;             // attention output    [B*T, C]

    cvt_kernel<<<2048, 256, 0, stream>>>(x, xb, (int)nBTC);
    cvt_kernel<<<1024, 256, 0, stream>>>(lw, lwb, Cc * Cc);
    cvtw_kernel<<<dim3(1024, 3), 256, 0, stream>>>(Wq, Wk, Wv, Wt);

    qkv_kernel<<<dim3((Bc * Tc) / 128, NHc, 3), 256, 0, stream>>>(xb, Wt, Qb, Kb, Vt);
    attn_kernel<<<dim3(Tc / 16, NHc, Bc), 32, 0, stream>>>(Qb, Kb, Vt, resb);
    linear_kernel<<<dim3((Bc * Tc) / 16, Cc / 64), 32, 0, stream>>>(resb, lwb, lb, out);
}
